// MultiheadAttention_62311385530901
// MI455X (gfx1250) — matrix operand order repaired and checked by reading, not compiled, not run on hardware
//
#include <hip/hip_runtime.h>
#include <hip/hip_bf16.h>

// ---------------------------------------------------------------------------
// Fused causal multi-head attention for MI455X (gfx1250, wave32, WMMA).
// B=2048, T=64, C=64, H=8, D=64.
// One block per batch element, one wave32 per head, everything fused so the
// only HBM traffic is x (fp32 in), out (fp32), and L2-resident f16 weights.
// Matrix math via v_wmma_f32_16x16x32_f16.
// ---------------------------------------------------------------------------

typedef _Float16 half_t;
typedef __attribute__((ext_vector_type(16))) _Float16 v16h;
typedef __attribute__((ext_vector_type(8)))  _Float16 v8h;
typedef __attribute__((ext_vector_type(8)))  float    v8f;

#define NB 2048
#define NT 64
#define NC 64
#define NH 8
#define ND 64
#define HD (NH * ND)   // 512

// ----- WMMA fragment loaders (wave32 layouts, ISA 7.12.2) -------------------
// A-matrix 16x32 f16: lane m=l&15 holds row m. VGPR slots 0..7 = K[8*half .. 8*half+7],
// slots 8..15 = K[16+8*half .. 23+8*half].
__device__ __forceinline__ v16h load_a_frag(const half_t* A, int lda, int row0, int k0) {
    const int l  = threadIdx.x & 31;
    const int m  = l & 15;
    const int hf = l >> 4;
    const half_t* p = A + (size_t)(row0 + m) * lda + k0 + 8 * hf;
    v8h lo = *(const v8h*)(p);
    v8h hi = *(const v8h*)(p + 16);
    v16h a;
#pragma unroll
    for (int i = 0; i < 8; ++i) { a[i] = lo[i]; a[i + 8] = hi[i]; }
    return a;
}

// B-matrix 32x16 f16, sourced from a K-contiguous ("B-transposed", N rows x K cols)
// buffer: lane n=l&15 holds column n; slots 0..7 = K[8*half .. 8*half+7], slots 8..15 = K[16+8*half .. 23+8*half].
__device__ __forceinline__ v16h load_b_frag(const half_t* BT, int ldb, int col0, int k0) {
    const int l  = threadIdx.x & 31;
    const int n  = l & 15;
    const int hf = l >> 4;
    const half_t* p = BT + (size_t)(col0 + n) * ldb + k0 + 8 * hf;
    v8h lo = *(const v8h*)(p);
    v8h hi = *(const v8h*)(p + 16);
    v16h b;
#pragma unroll
    for (int i = 0; i < 8; ++i) { b[i] = lo[i]; b[i + 8] = hi[i]; }
    return b;
}

// One 16x16 f32 output tile of C = A(MxK) * B(KxN), A row-major, B given K-contiguous.
template <int K>
__device__ __forceinline__ v8f mm_tile(const half_t* A, int lda,
                                       const half_t* BT, int ldb,
                                       int row0, int col0) {
    v8f acc = {};
#pragma unroll
    for (int k = 0; k < K; k += 32) {
        v16h a = load_a_frag(A, lda, row0, k);
        v16h b = load_b_frag(BT, ldb, col0, k);
        acc = __builtin_amdgcn_wmma_f32_16x16x32_f16(false, a, false, b,
                                                     (short)0, acc, false, false);
    }
    return acc;
}

// ----- Kernel 0: one-time weight convert + transpose to f16 in workspace ----
// WqT/WkT/WvT: [H][D][C]  (B-operand layout for q = x * Wq etc.)
// WpT:         [C][H*D]   (B-operand layout for out = concat * Wproj)
extern "C" __global__ __launch_bounds__(256)
void mha_prep_weights(const float* __restrict__ Wq, const float* __restrict__ Wk,
                      const float* __restrict__ Wv, const float* __restrict__ Wp,
                      half_t* __restrict__ wqt, half_t* __restrict__ wkt,
                      half_t* __restrict__ wvt, half_t* __restrict__ wpt) {
    int i = blockIdx.x * 256 + threadIdx.x;
    if (i < NH * NC * ND) {
        int h = i / (NC * ND), rem = i % (NC * ND);
        int d = rem / NC, c = rem % NC;
        int src = h * NC * ND + c * ND + d;       // Wq[h][c][d]
        wqt[i] = (half_t)Wq[src];
        wkt[i] = (half_t)Wk[src];
        wvt[i] = (half_t)Wv[src];
    }
    if (i < NC * HD) {
        int c = i / HD, k = i % HD;
        wpt[i] = (half_t)Wp[(size_t)k * NC + c]; // Wproj[k][c]
    }
}

// ----- Kernel 1: fused attention --------------------------------------------
extern "C" __global__ __launch_bounds__(256)
void mha_fused(const float* __restrict__ x,
               const half_t* __restrict__ wqt, const half_t* __restrict__ wkt,
               const half_t* __restrict__ wvt, const half_t* __restrict__ wpt,
               const float* __restrict__ bproj, float* __restrict__ out) {
    extern __shared__ char smem_raw[];
    half_t* xs    = (half_t*)smem_raw;           // [64][64]          8 KB
    half_t* heads = xs + NT * NC;                // 8 heads * 3 bufs  192 KB
    half_t* ocat  = heads + NH * 3 * (NT * ND);  // [64][512]         64 KB

    const int b   = blockIdx.x;
    const int tid = threadIdx.x;
    const int w   = tid >> 5;        // wave id == head id
    const int l   = tid & 31;
    const int n   = l & 15;          // N-lane within tile
    const int hf  = l >> 4;          // lane half (adds 8 to M, selects K halves)

    // --- stage x[b] as f16 in LDS (cooperative, coalesced) ---
    const float* xb = x + (size_t)b * (NT * NC);
    for (int i = tid; i < NT * NC; i += 256) xs[i] = (half_t)xb[i];
    __syncthreads();

    half_t* qbuf  = heads + w * 3 * (NT * ND);   // q, later reused for wei
    half_t* kbuf  = qbuf + NT * ND;              // k row-major [t][d]
    half_t* vbufT = kbuf + NT * ND;              // v transposed [d][t]
    const half_t* wq = wqt + w * (ND * NC);
    const half_t* wk = wkt + w * (ND * NC);
    const half_t* wv = wvt + w * (ND * NC);

    // --- QKV projections: 64x64x64 each, results staged in LDS ---
#pragma unroll
    for (int ti = 0; ti < 4; ++ti) {
#pragma unroll
        for (int j = 0; j < 4; ++j) {
            v8f aq = mm_tile<NC>(xs, NC, wq, NC, ti * 16, j * 16);
            v8f ak = mm_tile<NC>(xs, NC, wk, NC, ti * 16, j * 16);
            v8f av = mm_tile<NC>(xs, NC, wv, NC, ti * 16, j * 16);
#pragma unroll
            for (int r = 0; r < 8; ++r) {
                int t = ti * 16 + r + 8 * hf;
                int d = j * 16 + n;
                qbuf[t * ND + d]  = (half_t)aq[r];
                kbuf[t * ND + d]  = (half_t)ak[r];
                vbufT[d * NT + t] = (half_t)av[r];  // transpose for B-operand of P*V
            }
        }
    }

    // --- scores = q k^T * D^-0.5, all 16 tiles live in registers ---
    const float scale = 0.125f;  // 64^-0.5
    v8f s[4][4];
#pragma unroll
    for (int ti = 0; ti < 4; ++ti)
#pragma unroll
        for (int j = 0; j < 4; ++j)
            s[ti][j] = mm_tile<ND>(qbuf, ND, kbuf, ND, ti * 16, j * 16);

    // causal mask + scale
#pragma unroll
    for (int ti = 0; ti < 4; ++ti)
#pragma unroll
        for (int j = 0; j < 4; ++j)
#pragma unroll
            for (int r = 0; r < 8; ++r) {
                int t = ti * 16 + r + 8 * hf;
                int c = j * 16 + n;
                float v = s[ti][j][r] * scale;
                s[ti][j][r] = (c <= t) ? v : -__builtin_inff();
            }

    // row softmax: reduce over 4 column tiles (registers) + 16 N-lanes (shuffles)
#pragma unroll
    for (int ti = 0; ti < 4; ++ti) {
#pragma unroll
        for (int r = 0; r < 8; ++r) {
            float m = s[ti][0][r];
#pragma unroll
            for (int j = 1; j < 4; ++j) m = fmaxf(m, s[ti][j][r]);
#pragma unroll
            for (int off = 1; off < 16; off <<= 1)
                m = fmaxf(m, __shfl_xor(m, off, 32));
            float sum = 0.f;
#pragma unroll
            for (int j = 0; j < 4; ++j) {
                float e = __expf(s[ti][j][r] - m);
                s[ti][j][r] = e;
                sum += e;
            }
#pragma unroll
            for (int off = 1; off < 16; off <<= 1)
                sum += __shfl_xor(sum, off, 32);
            float inv = 1.0f / sum;
#pragma unroll
            for (int j = 0; j < 4; ++j) s[ti][j][r] *= inv;
        }
    }

    // --- wei -> f16 in LDS (reuse q buffer; q is dead) ---
#pragma unroll
    for (int ti = 0; ti < 4; ++ti)
#pragma unroll
        for (int j = 0; j < 4; ++j)
#pragma unroll
            for (int r = 0; r < 8; ++r) {
                int t = ti * 16 + r + 8 * hf;
                int c = j * 16 + n;
                qbuf[t * NT + c] = (half_t)s[ti][j][r];
            }

    // --- out_h = wei * v, written into the per-batch concat buffer ---
    half_t* och = ocat + w * ND;                  // columns [h*64 .. h*64+63], ld=512
#pragma unroll
    for (int ti = 0; ti < 4; ++ti)
#pragma unroll
        for (int j = 0; j < 4; ++j) {
            v8f o = mm_tile<NT>(qbuf, NT, vbufT, NT, ti * 16, j * 16);
#pragma unroll
            for (int r = 0; r < 8; ++r) {
                int t = ti * 16 + r + 8 * hf;
                och[t * HD + j * 16 + n] = (half_t)o[r];
            }
        }
    __syncthreads();

    // --- output projection: 64x512x64, 16 tiles split 2 per wave ---
    float* outb = out + (size_t)b * (NT * NC);
#pragma unroll
    for (int q = 0; q < 2; ++q) {
        int tt = 2 * w + q;
        int ti = tt >> 2, j = tt & 3;
        v8f acc = mm_tile<HD>(ocat, HD, wpt, HD, ti * 16, j * 16);
        float bias = bproj[j * 16 + n];
#pragma unroll
        for (int r = 0; r < 8; ++r) {
            int t = ti * 16 + r + 8 * hf;
            outb[t * NC + j * 16 + n] = acc[r] + bias;  // coalesced f32 stores
        }
    }
}

// ---------------------------------------------------------------------------
extern "C" void kernel_launch(void* const* d_in, const int* in_sizes, int n_in,
                              void* d_out, int out_size, void* d_ws, size_t ws_size,
                              hipStream_t stream) {
    const float* x     = (const float*)d_in[0];
    const float* Wq    = (const float*)d_in[1];
    const float* Wk    = (const float*)d_in[2];
    const float* Wv    = (const float*)d_in[3];
    const float* Wproj = (const float*)d_in[4];
    const float* bproj = (const float*)d_in[5];
    float* out = (float*)d_out;

    // Workspace: f16 transposed weights (4 * 32768 halfs = 256 KB).
    half_t* wqt = (half_t*)d_ws;
    half_t* wkt = wqt + NH * NC * ND;
    half_t* wvt = wkt + NH * NC * ND;
    half_t* wpt = wvt + NH * NC * ND;

    mha_prep_weights<<<(NC * HD + 255) / 256, 256, 0, stream>>>(
        Wq, Wk, Wv, Wproj, wqt, wkt, wvt, wpt);

    // LDS: x tile + 8*(q,k,vT) + concat = 8 + 192 + 64 KB = 264 KB (<320 KB/WG).
    const size_t lds_bytes =
        sizeof(half_t) * (NT * NC + NH * 3 * (NT * ND) + NT * HD);
    (void)hipFuncSetAttribute((const void*)mha_fused,
                              hipFuncAttributeMaxDynamicSharedMemorySize,
                              (int)lds_bytes);

    mha_fused<<<NB, 256, lds_bytes, stream>>>(x, wqt, wkt, wvt, wpt, bproj, out);
}
